// GraphNet_30915174596644
// MI455X (gfx1250) — compile-verified, matrix operand order checked by reading, not run on hardware
//
#include <hip/hip_runtime.h>

typedef __attribute__((ext_vector_type(16))) _Float16 v16h;
typedef __attribute__((ext_vector_type(8)))  _Float16 v8h;
typedef __attribute__((ext_vector_type(8)))  float    v8f;
typedef __attribute__((ext_vector_type(4)))  float    v4f;

#define BM 128
#define BN 128
#define BK 32
#define LDT 40   // padded LDS row stride (halves): 80B => conflict-free banks, 16B aligned

// Async DMA: 16 bytes per lane, global -> LDS, tracked by ASYNCcnt.
__device__ __forceinline__ void async_copy16(const _Float16* g, _Float16* l)
{
    const unsigned loff = (unsigned)(unsigned long long)l;   // flat low 32 bits == LDS byte addr
    asm volatile("global_load_async_to_lds_b128 %0, %1, off"
                 :: "v"(loff), "v"(g) : "memory");
}
__device__ __forceinline__ void wait_async0()
{
    asm volatile("s_wait_asynccnt 0x0" ::: "memory");
}

// ---------------------------------------------------------------------------
// f16 WMMA GEMM with async-LDS double buffering:
//   C[M,N] = act(A[M,K] @ Bt[N,K]^T + bias)
// Requirements: M % 128 == 0, K % 32 == 0, Bt padded with zero rows up to
// ceil(N/128)*128 so staging needs no guards. OUTH selects f16 vs f32 output.
// ---------------------------------------------------------------------------
template<bool OUTH>
__launch_bounds__(256)
__global__ void k_gemm_h16(const _Float16* __restrict__ A,
                           const _Float16* __restrict__ Bt,
                           const float* __restrict__ bias,
                           float* __restrict__ outF,
                           _Float16* __restrict__ outHp,
                           int M, int N, int K, int relu)
{
    __shared__ _Float16 sA[2][BM * LDT];
    __shared__ _Float16 sB[2][BN * LDT];

    const int t    = threadIdx.x;
    const int m0   = blockIdx.y * BM;
    const int n0   = blockIdx.x * BN;
    const int w    = t >> 5;
    const int lane = t & 31;
    const int wm   = w >> 1;            // 0..3 (32 rows each)
    const int wn   = w & 1;             // 0..1 (64 cols each)
    const int lr   = lane & 15;
    const int khalf = (lane >> 4) * 8;

    // staging geometry: each thread DMAs 2 chunks (16B) of A and 2 of B
    const int srow = t >> 2;            // 0..63
    const int skc  = (t & 3) * 8;       // 0,8,16,24

    union FH { v16h v; v8h h[2]; };

    v8f acc[2][4];
    for (int mi = 0; mi < 2; ++mi)
        for (int ni = 0; ni < 4; ++ni)
            for (int i = 0; i < 8; ++i) acc[mi][ni][i] = 0.0f;

    auto issue = [&](int k0, int b) {
        const _Float16* ga = A  + (size_t)(m0 + srow) * K + k0 + skc;
        const _Float16* gb = Bt + (size_t)(n0 + srow) * K + k0 + skc;
        _Float16* la = &sA[b][srow * LDT + skc];
        _Float16* lb = &sB[b][srow * LDT + skc];
        async_copy16(ga,                  la);
        async_copy16(ga + (size_t)64 * K, la + 64 * LDT);
        async_copy16(gb,                  lb);
        async_copy16(gb + (size_t)64 * K, lb + 64 * LDT);
    };

    issue(0, 0);
    int buf = 0;
    for (int k0 = 0; k0 < K; k0 += BK) {
        wait_async0();          // my DMAs for sA/sB[buf] are done
        __syncthreads();        // everyone's DMAs done; everyone done reading buf^1
        if (k0 + BK < K) issue(k0 + BK, buf ^ 1);   // prefetch next tile

        FH fa[2], fb[4];
        for (int mi = 0; mi < 2; ++mi) {
            const _Float16* p = &sA[buf][(wm * 32 + mi * 16 + lr) * LDT + khalf];
            fa[mi].h[0] = *(const v8h*)p;
            fa[mi].h[1] = *(const v8h*)(p + 16);
        }
        for (int ni = 0; ni < 4; ++ni) {
            const _Float16* p = &sB[buf][(wn * 64 + ni * 16 + lr) * LDT + khalf];
            fb[ni].h[0] = *(const v8h*)p;
            fb[ni].h[1] = *(const v8h*)(p + 16);
        }
        for (int mi = 0; mi < 2; ++mi)
            for (int ni = 0; ni < 4; ++ni)
                acc[mi][ni] = __builtin_amdgcn_wmma_f32_16x16x32_f16(
                    false, fa[mi].v, false, fb[ni].v,
                    (short)0, acc[mi][ni], false, false);
        buf ^= 1;
    }

    for (int mi = 0; mi < 2; ++mi) {
        const int mbase = m0 + wm * 32 + mi * 16 + (lane >> 4) * 8;
        for (int ni = 0; ni < 4; ++ni) {
            const int n = n0 + wn * 64 + ni * 16 + lr;
            if (n >= N) continue;
            const float bv = bias ? bias[n] : 0.0f;
            for (int r = 0; r < 8; ++r) {
                float v = acc[mi][ni][r] + bv;
                if (relu) v = v > 0.0f ? v : 0.0f;
                const size_t off = (size_t)(mbase + r) * N + n;
                if (OUTH) outHp[off] = (_Float16)v;
                else      outF[off]  = v;
            }
        }
    }
}

// ---------------------------------------------------------------------------
// fp32-A WMMA GEMM (A converted f32->f16 during LDS staging). Used only for
// P = new_edges(f32, in d_out) @ WPt. N == 128 exactly, no bias, no relu.
// ---------------------------------------------------------------------------
__launch_bounds__(256)
__global__ void k_gemm_af32(const float* __restrict__ A,
                            const _Float16* __restrict__ Bt,
                            float* __restrict__ outF,
                            int M, int N, int K)
{
    __shared__ _Float16 sA[BM * LDT];
    __shared__ _Float16 sB[BN * LDT];

    const int t    = threadIdx.x;
    const int m0   = blockIdx.y * BM;
    const int n0   = blockIdx.x * BN;
    const int w    = t >> 5;
    const int lane = t & 31;
    const int wm   = w >> 1;
    const int wn   = w & 1;
    const int lr   = lane & 15;
    const int khalf = (lane >> 4) * 8;

    union FH { v16h v; v8h h[2]; };

    v8f acc[2][4];
    for (int mi = 0; mi < 2; ++mi)
        for (int ni = 0; ni < 4; ++ni)
            for (int i = 0; i < 8; ++i) acc[mi][ni][i] = 0.0f;

    for (int k0 = 0; k0 < K; k0 += BK) {
        __syncthreads();
        for (int c = 0; c < 2; ++c) {
            const int row = (t >> 2) + 64 * c;
            const int kc  = (t & 3) * 8;
            v4f f0 = *(const v4f*)(A + (size_t)(m0 + row) * K + k0 + kc);
            v4f f1 = *(const v4f*)(A + (size_t)(m0 + row) * K + k0 + kc + 4);
            v8h av;
            for (int i = 0; i < 4; ++i) {
                av[i]     = (_Float16)f0[i];
                av[i + 4] = (_Float16)f1[i];
            }
            *(v8h*)(sA + row * LDT + kc) = av;
            *(v8h*)(sB + row * LDT + kc) =
                *(const v8h*)(Bt + (size_t)(n0 + row) * K + k0 + kc);
        }
        __syncthreads();

        FH fa[2], fb[4];
        for (int mi = 0; mi < 2; ++mi) {
            const _Float16* p = sA + (wm * 32 + mi * 16 + lr) * LDT + khalf;
            fa[mi].h[0] = *(const v8h*)p;
            fa[mi].h[1] = *(const v8h*)(p + 16);
        }
        for (int ni = 0; ni < 4; ++ni) {
            const _Float16* p = sB + (wn * 64 + ni * 16 + lr) * LDT + khalf;
            fb[ni].h[0] = *(const v8h*)p;
            fb[ni].h[1] = *(const v8h*)(p + 16);
        }
        for (int mi = 0; mi < 2; ++mi)
            for (int ni = 0; ni < 4; ++ni)
                acc[mi][ni] = __builtin_amdgcn_wmma_f32_16x16x32_f16(
                    false, fa[mi].v, false, fb[ni].v,
                    (short)0, acc[mi][ni], false, false);
    }

    for (int mi = 0; mi < 2; ++mi) {
        const int mbase = m0 + wm * 32 + mi * 16 + (lane >> 4) * 8;
        for (int ni = 0; ni < 4; ++ni) {
            const int n = n0 + wn * 64 + ni * 16 + lr;
            for (int r = 0; r < 8; ++r)
                outF[(size_t)(mbase + r) * N + n] = acc[mi][ni][r];
        }
    }
}

// ---------------------------------------------------------------------------
// Transpose + f32->f16 convert:  Wt[n*ldt + k] = (k<K && n<N ? W[k*ldw+n] : 0)
// writes all n in [0, Npad), k in [0, Kpad)  (zero padding for guard-free GEMM)
// ---------------------------------------------------------------------------
__launch_bounds__(256)
__global__ void k_transpose_f16(const float* __restrict__ W, int K, int N, int ldw,
                                _Float16* __restrict__ Wt, int Kpad, int ldt, int Npad)
{
    __shared__ float tile[32][33];
    const int tx = threadIdx.x, ty = threadIdx.y;   // 32 x 8
    const int k0 = blockIdx.x * 32;
    const int n0 = blockIdx.y * 32;
    for (int r = 0; r < 4; ++r) {
        const int k = k0 + ty + 8 * r;
        const int n = n0 + tx;
        tile[ty + 8 * r][tx] = (k < K && n < N) ? W[(size_t)k * ldw + n] : 0.0f;
    }
    __syncthreads();
    for (int r = 0; r < 4; ++r) {
        const int n = n0 + ty + 8 * r;
        const int k = k0 + tx;
        if (n < Npad && k < Kpad)
            Wt[(size_t)n * ldt + k] = (_Float16)tile[tx][ty + 8 * r];
    }
}

// e_in[4096][160] f16: [edges(16) | nodes[s](64) | nodes[r](64) | g(8) | 0(8)]
__global__ void k_pack_ein(const float* __restrict__ edges,
                           const float* __restrict__ nodes,
                           const float* __restrict__ g,
                           const int* __restrict__ senders,
                           const int* __restrict__ receivers,
                           _Float16* __restrict__ ein)
{
    const int idx = blockIdx.x * blockDim.x + threadIdx.x;
    const int e = idx / 160;
    const int k = idx - e * 160;
    float v;
    if      (k < 16)  v = edges[e * 16 + k];
    else if (k < 80)  v = nodes[(size_t)senders[e]   * 64 + (k - 16)];
    else if (k < 144) v = nodes[(size_t)receivers[e] * 64 + (k - 80)];
    else if (k < 152) v = g[k - 144];
    else              v = 0.0f;
    ein[idx] = (_Float16)v;
}

// acc[i][j] = nb1[j] + nodes[i,:]@nW1[0:64,j] + g@nW1[8256:8264,j]
__global__ void k_node_init(const float* __restrict__ nodes,
                            const float* __restrict__ g,
                            const float* __restrict__ nW1,
                            const float* __restrict__ nb1,
                            float* __restrict__ acc)
{
    const int idx = blockIdx.x * blockDim.x + threadIdx.x;
    const int i = idx >> 6;
    const int j = idx & 63;
    float s = nb1[j];
    for (int k = 0; k < 64; ++k) s += nodes[(size_t)i * 64 + k] * nW1[(size_t)k * 64 + j];
    for (int k = 0; k < 8;  ++k) s += g[k] * nW1[(size_t)(8256 + k) * 64 + j];
    acc[idx] = s;
}

// scatter-add P rows into node accumulator (segment_sum commuted past weights)
__global__ void k_scatter(const float* __restrict__ P,
                          const int* __restrict__ senders,
                          const int* __restrict__ receivers,
                          float* __restrict__ acc)
{
    const int idx = blockIdx.x * blockDim.x + threadIdx.x;
    const int e = idx >> 7;
    const int j = idx & 127;
    const float v = P[idx];
    if (j < 64) atomicAdd(&acc[(size_t)senders[e]   * 64 + j],        v);
    else        atomicAdd(&acc[(size_t)receivers[e] * 64 + (j - 64)], v);
}

__global__ void k_relu_h2(const float* __restrict__ acc, _Float16* __restrict__ h2)
{
    const int idx = blockIdx.x * blockDim.x + threadIdx.x;
    const float v = acc[idx];
    h2[idx] = (_Float16)(v > 0.0f ? v : 0.0f);
}

extern "C" void kernel_launch(void* const* d_in, const int* in_sizes, int n_in,
                              void* d_out, int out_size, void* d_ws, size_t ws_size,
                              hipStream_t stream)
{
    const float* nodes   = (const float*)d_in[0];
    const float* edges   = (const float*)d_in[1];
    const float* g       = (const float*)d_in[2];
    const int*   senders = (const int*)d_in[3];
    const int*   recvs   = (const int*)d_in[4];
    const float* eW1 = (const float*)d_in[5];
    const float* eb1 = (const float*)d_in[6];
    const float* eW2 = (const float*)d_in[7];
    const float* eb2 = (const float*)d_in[8];
    const float* nW1 = (const float*)d_in[9];
    const float* nb1 = (const float*)d_in[10];
    const float* nW2 = (const float*)d_in[11];
    const float* nb2 = (const float*)d_in[12];

    float* out_nodes = (float*)d_out;                       // [8192,64]
    float* out_edges = (float*)d_out + (size_t)8192 * 64;   // [4096,4096]

    char* ws = (char*)d_ws;
    size_t o = 0;
    auto take = [&](size_t bytes) -> char* {
        char* p = ws + o;
        o = (o + bytes + 255) & ~(size_t)255;
        return p;
    };
    _Float16* ein   = (_Float16*)take((size_t)4096 * 160 * 2);
    _Float16* eW1t  = (_Float16*)take((size_t)4096 * 160 * 2);   // [4096,160]
    _Float16* hbuf  = (_Float16*)take((size_t)4096 * 4096 * 2);  // h (f16)
    _Float16* eW2t  = (_Float16*)take((size_t)4096 * 4096 * 2);  // [4096,4096]
    _Float16* WPt   = (_Float16*)take((size_t)128 * 4096 * 2);   // [128,4096]
    float*    P     = (float*)   take((size_t)4096 * 128 * 4);   // [4096,128]
    float*    acc   = (float*)   take((size_t)8192 * 64 * 4);
    _Float16* h2    = (_Float16*)take((size_t)8192 * 64 * 2);
    _Float16* nW2t  = (_Float16*)take((size_t)128 * 64 * 2);     // padded to 128 rows
    (void)ws_size;

    const dim3 tb(32, 8);
    // ---- weight transpose + convert (Bt buffers padded to >= ceil(N/128)*128 rows) ----
    k_transpose_f16<<<dim3(5, 128),   tb, 0, stream>>>(eW1, 152, 4096, 4096, eW1t, 160, 160, 4096);
    k_transpose_f16<<<dim3(128, 128), tb, 0, stream>>>(eW2, 4096, 4096, 4096, eW2t, 4096, 4096, 4096);
    k_transpose_f16<<<dim3(128, 2),   tb, 0, stream>>>(nW1 + (size_t)64 * 64,   4096, 64, 64, WPt,                      4096, 4096, 64);
    k_transpose_f16<<<dim3(128, 2),   tb, 0, stream>>>(nW1 + (size_t)4160 * 64, 4096, 64, 64, WPt + (size_t)64 * 4096,  4096, 4096, 64);
    k_transpose_f16<<<dim3(2, 4),     tb, 0, stream>>>(nW2, 64, 64, 64, nW2t, 64, 64, 128);

    // ---- edge MLP ----
    k_pack_ein<<<(4096 * 160) / 256, 256, 0, stream>>>(edges, nodes, g, senders, recvs, ein);
    k_gemm_h16<true ><<<dim3(32, 32), 256, 0, stream>>>(ein,  eW1t, eb1, nullptr,   hbuf,    4096, 4096, 160,  1);
    k_gemm_h16<false><<<dim3(32, 32), 256, 0, stream>>>(hbuf, eW2t, eb2, out_edges, nullptr, 4096, 4096, 4096, 1);

    // ---- node path ----
    k_node_init<<<(8192 * 64) / 256, 256, 0, stream>>>(nodes, g, nW1, nb1, acc);
    k_gemm_af32<<<dim3(1, 32), 256, 0, stream>>>(out_edges, WPt, P, 4096, 128, 4096);
    k_scatter<<<(4096 * 128) / 256, 256, 0, stream>>>(P, senders, recvs, acc);
    k_relu_h2<<<(8192 * 64) / 256, 256, 0, stream>>>(acc, h2);
    k_gemm_h16<false><<<dim3(1, 64), 256, 0, stream>>>(h2, nW2t, nb2, out_nodes, nullptr, 8192, 64, 64, 1);
}
